// RaggedAttention_72095321031178
// MI455X (gfx1250) — compile-verified, matrix operand order checked by reading, not run on hardware
//
#include <hip/hip_runtime.h>
#include <hip/hip_bf16.h>

// ---------------------------------------------------------------------------
// Ragged (pad-packed) attention for MI455X / gfx1250.
// All GEMM-like math runs on V_WMMA_F32_16X16X32_BF16 (wave32 WMMA).
// V projection stores V^T so attention AV B-fragments are contiguous loads.
// ---------------------------------------------------------------------------

#define BB 8
#define TT 1024
#define CC 1024
#define HH 16
#define HD 64
#define TOTAL (BB * TT)      // 8192 tokens
#define NEGV  (-1e9f)

typedef __attribute__((ext_vector_type(16))) __bf16 v16bf;
typedef __attribute__((ext_vector_type(8)))  float  v8f;

__device__ __forceinline__ v8f wmma_bf16(v16bf a, v16bf b, v8f c) {
  // (neg_a, A, neg_b, B, c_mod, C, reuse_a, reuse_b)
  return __builtin_amdgcn_wmma_f32_16x16x32_bf16(false, a, false, b,
                                                 (short)0, c, false, false);
}

// K-offset pattern for 16-bit 16x32 A fragments (ISA 7.12.2):
//   VGPR v (0..7), lane-half hi: pair base = (v<4?0:16) + hi*8 + 2*(v&3)
__device__ __forceinline__ int frag_kbase(int v, int hi) {
  return ((v & 4) << 2) + (hi << 3) + ((v & 3) << 1);
}

// Row-major fragment loader: element(kb) = p[row*ld + k0 + kb].
// Used for A fragments (rows of activations / P tile), B fragments whose
// matrix column is contiguous in memory (W rows for y = x W^T, K rows for
// Q K^T), and B fragments of V^T (channel-major V).
__device__ __forceinline__ v16bf load_frag_rm(const __bf16* __restrict__ p,
                                              int ld, int row, int k0, int hi) {
  union { v16bf v; unsigned u[8]; } r;
  const unsigned* base = reinterpret_cast<const unsigned*>(p + row * ld + k0);
  #pragma unroll
  for (int v = 0; v < 8; ++v) {
    r.u[v] = base[frag_kbase(v, hi) >> 1];   // aligned bf16 pair
  }
  return r.v;
}

// ---------------------------------------------------------------------------
// Conversion kernels
// ---------------------------------------------------------------------------
__global__ __launch_bounds__(256) void cvt_gather_x(const float* __restrict__ x,
                                                    const int* __restrict__ idx,
                                                    __bf16* __restrict__ xb) {
  int i = blockIdx.x * 256 + threadIdx.x;           // over TOTAL*CC
  int t = i >> 10, c = i & (CC - 1);
  xb[i] = (__bf16)x[idx[t] * CC + c];
}

__global__ __launch_bounds__(256) void cvt_w(const float* __restrict__ w,
                                             __bf16* __restrict__ wb) {
  int i = blockIdx.x * 256 + threadIdx.x;           // over CC*CC
  wb[i] = (__bf16)w[i];
}

// ---------------------------------------------------------------------------
// GEMM:  acc[m, n] = sum_k A[m,k] * W[n,k] + bias[n]   (torch Linear)
// One wave -> 16x64 output tile; K loop in steps of 32; 4 WMMAs per step.
// TR=false: out[m*CC + n]   (row-major, OutT)
// TR=true : out[n*M  + m]   (transposed, used to produce V^T)
// ---------------------------------------------------------------------------
template <typename OutT, bool TR>
__global__ __launch_bounds__(128) void gemm_wt(const __bf16* __restrict__ A,
                                               const __bf16* __restrict__ W,
                                               const float* __restrict__ bias,
                                               OutT* __restrict__ out, int M) {
  const int lane = threadIdx.x & 31;
  const int wid  = blockIdx.x * 4 + (threadIdx.x >> 5);
  const int nTN  = CC / 64;                      // 16 N-tiles
  const int m0 = (wid / nTN) * 16;
  const int n0 = (wid % nTN) * 64;
  if (m0 >= M) return;
  const int hi = lane >> 4, li = lane & 15;

  v8f acc0 = {}, acc1 = {}, acc2 = {}, acc3 = {};
  for (int k0 = 0; k0 < CC; k0 += 32) {
    v16bf a  = load_frag_rm(A, CC, m0 + li, k0, hi);
    v16bf b0 = load_frag_rm(W, CC, n0 + li,      k0, hi);
    v16bf b1 = load_frag_rm(W, CC, n0 + 16 + li, k0, hi);
    v16bf b2 = load_frag_rm(W, CC, n0 + 32 + li, k0, hi);
    v16bf b3 = load_frag_rm(W, CC, n0 + 48 + li, k0, hi);
    acc0 = wmma_bf16(a, b0, acc0);
    acc1 = wmma_bf16(a, b1, acc1);
    acc2 = wmma_bf16(a, b2, acc2);
    acc3 = wmma_bf16(a, b3, acc3);
  }
  // bias + store. C/D layout: element (M = r + 8*hi, N = li) in VGPR r.
  v8f* accs[4] = {&acc0, &acc1, &acc2, &acc3};
  #pragma unroll
  for (int t = 0; t < 4; ++t) {
    const int n = n0 + t * 16 + li;
    float bn = bias[n];
    #pragma unroll
    for (int r = 0; r < 8; ++r) {
      int row = m0 + r + 8 * hi;
      if (TR) out[n * M + row] = (OutT)((*accs[t])[r] + bn);   // 8 contiguous
      else    out[row * CC + n] = (OutT)((*accs[t])[r] + bn);
    }
  }
}

// ---------------------------------------------------------------------------
// Flash attention: one wave handles one (b, h, 16-row i-tile).
// j loop in steps of 32: 4 score WMMAs, online softmax, LDS transpose of P,
// 4 AV WMMAs into a 16x64 f32 accumulator. V is channel-major (V^T).
// ---------------------------------------------------------------------------
__global__ __launch_bounds__(128) void attn_kernel(const __bf16* __restrict__ q,
                                                   const __bf16* __restrict__ k,
                                                   const __bf16* __restrict__ vT,
                                                   const int* __restrict__ pb,
                                                   __bf16* __restrict__ y) {
  __shared__ __align__(16) __bf16 lds[4][16 * 32];   // per-wave P tile
  const int lane = threadIdx.x & 31;
  const int wv   = threadIdx.x >> 5;
  const int wid  = blockIdx.x * 4 + wv;
  if (wid >= BB * HH * (TT / 16)) return;
  const int b  = wid >> 10;                // HH * (TT/16) = 1024
  const int rm = wid & 1023;
  const int h  = rm >> 6;
  const int i0 = (rm & 63) << 4;
  const int hi = lane >> 4, li = lane & 15;
  const float scale = 0.125f;              // 1/sqrt(64)

  const __bf16* qh = q + h * HD;
  const __bf16* kh = k + h * HD;

  // Q tile (16 x 64) as two A fragments (K=0..31, 32..63)
  v16bf aq0 = load_frag_rm(qh, CC, b * TT + i0 + li, 0, hi);
  v16bf aq1 = load_frag_rm(qh, CC, b * TT + i0 + li, 32, hi);

  int sid_i[8];
  #pragma unroll
  for (int r = 0; r < 8; ++r) sid_i[r] = pb[b * TT + i0 + r + 8 * hi];

  float mrow[8], lrow[8];
  #pragma unroll
  for (int r = 0; r < 8; ++r) { mrow[r] = -1e30f; lrow[r] = 0.0f; }
  v8f acc0 = {}, acc1 = {}, acc2 = {}, acc3 = {};

  for (int j0 = 0; j0 < TT; j0 += 32) {
    // prefetch next j-tile streams (global_prefetch_b8 on gfx1250)
    if (j0 + 32 < TT) {
      __builtin_prefetch(kh + (size_t)(b * TT + j0 + 32 + li) * CC, 0, 1);
      __builtin_prefetch(kh + (size_t)(b * TT + j0 + 48 + li) * CC, 0, 1);
      __builtin_prefetch(vT + (size_t)(h * HD + li) * TOTAL + b * TT + j0 + 32, 0, 1);
      __builtin_prefetch(vT + (size_t)(h * HD + 32 + li) * TOTAL + b * TT + j0 + 32, 0, 1);
    }

    // --- scores S = Q K^T for a 16x32 block of columns ---
    v8f sLo = {}, sHi = {};
    {
      v16bf bk;
      bk = load_frag_rm(kh, CC, b * TT + j0 + li, 0, hi);       sLo = wmma_bf16(aq0, bk, sLo);
      bk = load_frag_rm(kh, CC, b * TT + j0 + li, 32, hi);      sLo = wmma_bf16(aq1, bk, sLo);
      bk = load_frag_rm(kh, CC, b * TT + j0 + 16 + li, 0, hi);  sHi = wmma_bf16(aq0, bk, sHi);
      bk = load_frag_rm(kh, CC, b * TT + j0 + 16 + li, 32, hi); sHi = wmma_bf16(aq1, bk, sHi);
    }
    const int sjLo = pb[b * TT + j0 + li];
    const int sjHi = pb[b * TT + j0 + 16 + li];

    // --- mask, online softmax (row reductions across 16 lanes) ---
    #pragma unroll
    for (int r = 0; r < 8; ++r) {
      float a0 = (sjLo == sid_i[r]) ? sLo[r] * scale : NEGV;
      float a1 = (sjHi == sid_i[r]) ? sHi[r] * scale : NEGV;
      float tmax = fmaxf(a0, a1);
      #pragma unroll
      for (int mk = 1; mk < 16; mk <<= 1)
        tmax = fmaxf(tmax, __shfl_xor(tmax, mk, 16));
      float mnew = fmaxf(mrow[r], tmax);
      float cf   = __expf(mrow[r] - mnew);
      mrow[r]    = mnew;
      float e0 = __expf(a0 - mnew);
      float e1 = __expf(a1 - mnew);
      float s  = e0 + e1;
      #pragma unroll
      for (int mk = 1; mk < 16; mk <<= 1)
        s += __shfl_xor(s, mk, 16);
      lrow[r] = lrow[r] * cf + s;
      acc0[r] *= cf; acc1[r] *= cf; acc2[r] *= cf; acc3[r] *= cf;
      // stash P (bf16) row-major in LDS: element (M = r+8*hi, K = li / 16+li)
      lds[wv][(r + 8 * hi) * 32 + li]      = (__bf16)e0;
      lds[wv][(r + 8 * hi) * 32 + 16 + li] = (__bf16)e1;
    }
    // lane X wrote LDS that lane Y reads: compiler can't see it -> wait DScnt
    asm volatile("s_wait_dscnt 0x0" ::: "memory");

    // --- P as A fragment (16x32); V^T rows as B fragments; 4 AV WMMAs ---
    v16bf ap  = load_frag_rm(&lds[wv][0], 32, li, 0, hi);
    v16bf bv0 = load_frag_rm(vT, TOTAL, h * HD + 0 + li,  b * TT + j0, hi);
    v16bf bv1 = load_frag_rm(vT, TOTAL, h * HD + 16 + li, b * TT + j0, hi);
    v16bf bv2 = load_frag_rm(vT, TOTAL, h * HD + 32 + li, b * TT + j0, hi);
    v16bf bv3 = load_frag_rm(vT, TOTAL, h * HD + 48 + li, b * TT + j0, hi);
    acc0 = wmma_bf16(ap, bv0, acc0);
    acc1 = wmma_bf16(ap, bv1, acc1);
    acc2 = wmma_bf16(ap, bv2, acc2);
    acc3 = wmma_bf16(ap, bv3, acc3);
  }

  // --- epilogue: normalize and store y tile (bf16) ---
  v8f* accs[4] = {&acc0, &acc1, &acc2, &acc3};
  #pragma unroll
  for (int t = 0; t < 4; ++t) {
    #pragma unroll
    for (int r = 0; r < 8; ++r) {
      int row = b * TT + i0 + r + 8 * hi;
      y[row * CC + h * HD + t * 16 + li] = (__bf16)((*accs[t])[r] / lrow[r]);
    }
  }
}

// ---------------------------------------------------------------------------
// Scatter back to ragged layout: out[i] = yf[inverse_index[i]]
// ---------------------------------------------------------------------------
__global__ __launch_bounds__(256) void scatter_out(const float* __restrict__ yf,
                                                   const int* __restrict__ inv,
                                                   float* __restrict__ out) {
  int i = blockIdx.x * 256 + threadIdx.x;       // over TOTAL*CC
  int t = i >> 10, c = i & (CC - 1);
  out[i] = yf[inv[t] * CC + c];
}

// ---------------------------------------------------------------------------
extern "C" void kernel_launch(void* const* d_in, const int* in_sizes, int n_in,
                              void* d_out, int out_size, void* d_ws, size_t ws_size,
                              hipStream_t stream) {
  const float* x   = (const float*)d_in[0];
  const int*   ppi = (const int*)d_in[1];    // padpack_index (identity)
  const int*   ppb = (const int*)d_in[2];    // padpack_batch (seq ids)
  const int*   ppv = (const int*)d_in[3];    // padpack_inverse_index
  const float* Wq  = (const float*)d_in[4];
  const float* bq  = (const float*)d_in[5];
  const float* Wk  = (const float*)d_in[6];
  const float* bk  = (const float*)d_in[7];
  const float* Wv  = (const float*)d_in[8];
  const float* bv  = (const float*)d_in[9];
  const float* Wp  = (const float*)d_in[10];
  const float* bp  = (const float*)d_in[11];
  float* out = (float*)d_out;

  char* ws = (char*)d_ws;
  const size_t MB = 1024 * 1024;
  __bf16* xb  = (__bf16*)(ws + 0);          // 16 MB  [TOTAL, CC]
  __bf16* wqb = (__bf16*)(ws + 16 * MB);    //  2 MB
  __bf16* wkb = (__bf16*)(ws + 18 * MB);
  __bf16* wvb = (__bf16*)(ws + 20 * MB);
  __bf16* wpb = (__bf16*)(ws + 22 * MB);
  __bf16* qb  = (__bf16*)(ws + 24 * MB);    // 16 MB  [TOTAL, CC]
  __bf16* kb2 = (__bf16*)(ws + 40 * MB);    // 16 MB  [TOTAL, CC]
  __bf16* vTb = (__bf16*)(ws + 56 * MB);    // 16 MB  [CC, TOTAL] (V^T)
  __bf16* yb  = xb;                          // attention out reuses xb
  float*  yf  = (float*)(ws + 24 * MB);      // final proj f32 reuses q/k (32 MB)

  const int elemBlocks = (TOTAL * CC) / 256;        // 32768
  const int wBlocks    = (CC * CC) / 256;           // 4096
  const int gemmBlocks = ((TOTAL / 16) * (CC / 64)) / 4;   // 2048 (4 waves/blk)
  const int attnBlocks = (BB * HH * (TT / 16)) / 4;        // 2048

  // 1) convert inputs to bf16 (x gathered through padpack_index)
  cvt_gather_x<<<elemBlocks, 256, 0, stream>>>(x, ppi, xb);
  cvt_w<<<wBlocks, 256, 0, stream>>>(Wq, wqb);
  cvt_w<<<wBlocks, 256, 0, stream>>>(Wk, wkb);
  cvt_w<<<wBlocks, 256, 0, stream>>>(Wv, wvb);
  cvt_w<<<wBlocks, 256, 0, stream>>>(Wp, wpb);

  // 2) Q/K projections row-major; V projection stored transposed (V^T)
  gemm_wt<__bf16, false><<<gemmBlocks, 128, 0, stream>>>(xb, wqb, bq, qb, TOTAL);
  gemm_wt<__bf16, false><<<gemmBlocks, 128, 0, stream>>>(xb, wkb, bk, kb2, TOTAL);
  gemm_wt<__bf16, true ><<<gemmBlocks, 128, 0, stream>>>(xb, wvb, bv, vTb, TOTAL);

  // 3) masked flash attention (writes yb == xb, safe: xb no longer needed)
  attn_kernel<<<attnBlocks, 128, 0, stream>>>(qb, kb2, vTb, ppb, yb);

  // 4) output projection to f32, then scatter to ragged layout
  gemm_wt<float, false><<<gemmBlocks, 128, 0, stream>>>(yb, wpb, bp, yf, TOTAL);
  scatter_out<<<elemBlocks, 256, 0, stream>>>(yf, ppv, out);
}
